// ACT_90915867722334
// MI455X (gfx1250) — matrix operand order repaired and checked by reading, not compiled, not run on hardware
//
#include <hip/hip_runtime.h>

#define BATCH 8192
#define IN    64
#define HID   1024
#define OMID  128
#define KA    1088   /* IN + HID */
#define MAX_ITER 8
#define EPS 0.001f

typedef __attribute__((ext_vector_type(16))) __bf16 v16bf;
typedef __attribute__((ext_vector_type(8)))  __bf16 v8bf;
typedef __attribute__((ext_vector_type(8)))  float  v8f;

__device__ __forceinline__ unsigned short f2bf(float f) {
    unsigned u = __float_as_uint(f);
    u += 0x7FFFu + ((u >> 16) & 1u);          // round-to-nearest-even
    return (unsigned short)(u >> 16);
}
__device__ __forceinline__ float bf2f(unsigned short h) {
    return __uint_as_float(((unsigned)h) << 16);
}
__device__ __forceinline__ float sigm(float x) {
    return 1.0f / (1.0f + __expf(-x));
}
// branch-free tanh: clamp then (e^2x-1)/(e^2x+1) on the v_exp_f32 unit
__device__ __forceinline__ float fast_tanh(float x) {
    x = fminf(fmaxf(x, -15.0f), 15.0f);
    const float e = __expf(2.0f * x);
    return (e - 1.0f) / (e + 1.0f);
}

// ---- WMMA fragment loaders (contiguous-K order, same in A and B, wave32) -------------------
// A 16x32 bf16: lane L holds row M=L&15; 16 contiguous K values at offset (L>>4)*16.
__device__ __forceinline__ v16bf load_a_ptr(const unsigned short* p) {
    v8bf lo = *(const v8bf*)p;
    v8bf hi = *(const v8bf*)(p + 8);
    v16bf a;
#pragma unroll
    for (int i = 0; i < 8; ++i) { a[i] = lo[i]; a[i + 8] = hi[i]; }
    return a;
}
__device__ __forceinline__ v16bf load_a_frag(const unsigned short* __restrict__ base,
                                             int row0, int ld, int k, int lane) {
    const int r = lane & 15, h = lane >> 4;
    return load_a_ptr(base + (size_t)(row0 + r) * ld + k + h * 16);
}
// B 32x16 bf16 (B = W^T, W row-major [N][K]): lane L holds column N=L&15,
// 16 contiguous K values at offset (L>>4)*16.
__device__ __forceinline__ v16bf load_b_frag(const unsigned short* __restrict__ W,
                                             int nrow0, int ld, int k, int lane) {
    const int c = lane & 15, h = lane >> 4;
    return *(const v16bf*)(W + (size_t)(nrow0 + c) * ld + k + h * 16);
}

#define WMMA_BF16(A, B, C) \
    __builtin_amdgcn_wmma_f32_16x16x32_bf16(false, (A), false, (B), (short)0, (C), false, false)

// ---------------------------------------------------------------------------
// Fused 4-gate GEMM + LSTM cell/state update.
// Block: 256 thr = 8 waves as 2(row)x4(col). Block tile 128 rows x 64 hidden.
// Wave tile: 64 rows x 16 hidden x 4 gates -> 16 WMMA per K-step.
// State part of K is software-pipelined 2 deep (ping-pong fragment buffers).
// ---------------------------------------------------------------------------
__global__ __launch_bounds__(256) void k_gates_lstm(
    const unsigned short* __restrict__ Xbf,    // [BATCH][IN] bf16
    const unsigned short* __restrict__ Sread,  // [BATCH][HID] bf16 (state in)
    unsigned short* __restrict__ Swrite,       // [BATCH][HID] bf16 (state out)
    const unsigned short* __restrict__ Wcat,   // [4*HID][KA] bf16
    const float* __restrict__ bcat,            // [4*HID]
    float* __restrict__ Cell,                  // [BATCH][HID] f32
    const int* __restrict__ active)            // [BATCH]
{
    const int lane = threadIdx.x & 31;
    const int wave = threadIdx.x >> 5;
    const int m_base = blockIdx.y * 128 + (wave >> 2) * 64;
    const int n_base = blockIdx.x * 64  + (wave & 3) * 16;
    const int col  = n_base + (lane & 15);
    const int half = lane >> 4;
    const int r    = lane & 15;

    v8f acc[4][4];
#pragma unroll
    for (int g = 0; g < 4; ++g) {
        const float b = bcat[g * HID + col];
#pragma unroll
        for (int mt = 0; mt < 4; ++mt) {
#pragma unroll
            for (int i = 0; i < 8; ++i) acc[g][mt][i] = b;
        }
    }

    // ---- K part 1: x contribution (k = 0, 32) ----
#pragma unroll
    for (int k0 = 0; k0 < IN; k0 += 32) {
        v16bf ax[4];
#pragma unroll
        for (int mt = 0; mt < 4; ++mt)
            ax[mt] = load_a_frag(Xbf, m_base + mt * 16, IN, k0, lane);
#pragma unroll
        for (int g = 0; g < 4; ++g) {
            v16bf bx = load_b_frag(Wcat, g * HID + n_base, KA, k0, lane);
#pragma unroll
            for (int mt = 0; mt < 4; ++mt)
                acc[g][mt] = WMMA_BF16(ax[mt], bx, acc[g][mt]);
        }
    }

    // ---- K part 2: state contribution, 32 steps, pipelined 2x ----
    const unsigned short* pA[4];
#pragma unroll
    for (int mt = 0; mt < 4; ++mt)
        pA[mt] = Sread + (size_t)(m_base + mt * 16 + r) * HID + half * 16;
    const unsigned short* pB[4];
#pragma unroll
    for (int g = 0; g < 4; ++g)
        pB[g] = Wcat + (size_t)(g * HID + n_base + r) * KA + IN + half * 16;

    v16bf a0[4], b0[4], a1[4], b1[4];
#pragma unroll
    for (int mt = 0; mt < 4; ++mt) { a0[mt] = load_a_ptr(pA[mt]); pA[mt] += 32; }
#pragma unroll
    for (int g = 0; g < 4; ++g)    { b0[g] = *(const v16bf*)pB[g]; pB[g] += 32; }

#pragma unroll 1
    for (int k = 0; k < HID; k += 64) {
        // prefetch step k+32 (always in range: last k = 960 -> 992 < 1024)
#pragma unroll
        for (int mt = 0; mt < 4; ++mt) { a1[mt] = load_a_ptr(pA[mt]); pA[mt] += 32; }
#pragma unroll
        for (int g = 0; g < 4; ++g)    { b1[g] = *(const v16bf*)pB[g]; pB[g] += 32; }
        // math on step k
#pragma unroll
        for (int g = 0; g < 4; ++g)
#pragma unroll
            for (int mt = 0; mt < 4; ++mt)
                acc[g][mt] = WMMA_BF16(a0[mt], b0[g], acc[g][mt]);
        // prefetch step k+64
        if (k + 64 < HID) {
#pragma unroll
            for (int mt = 0; mt < 4; ++mt) { a0[mt] = load_a_ptr(pA[mt]); pA[mt] += 32; }
#pragma unroll
            for (int g = 0; g < 4; ++g)    { b0[g] = *(const v16bf*)pB[g]; pB[g] += 32; }
        }
        // math on step k+32
#pragma unroll
        for (int g = 0; g < 4; ++g)
#pragma unroll
            for (int mt = 0; mt < 4; ++mt)
                acc[g][mt] = WMMA_BF16(a1[mt], b1[g], acc[g][mt]);
    }

    // ---- Fused LSTM update epilogue (C/D layout: M = v + 8*half, N = lane&15).
#pragma unroll
    for (int mt = 0; mt < 4; ++mt) {
#pragma unroll
        for (int v = 0; v < 8; ++v) {
            const int row = m_base + mt * 16 + half * 8 + v;
            const float ig = sigm(acc[0][mt][v]);
            const float fg = sigm(acc[1][mt][v]);
            const float cg = fast_tanh(acc[2][mt][v]);
            const float og = sigm(acc[3][mt][v]);
            const size_t si = (size_t)row * HID + col;
            const unsigned short olds = Sread[si];
            unsigned short outs = olds;
            if (active[row]) {
                float cell = fg * Cell[si] + ig * cg;
                Cell[si] = cell;
                outs = f2bf(og * fast_tanh(cell));
            }
            Swrite[si] = outs;
        }
    }
}

// ---------------------------------------------------------------------------
// Head GEMM: M = relu(state @ W1^T + b1)  [BATCH x OMID], pipelined 2x.
// Block tile 128 rows x 64 cols, wave tile 64x16, K = HID.
// ---------------------------------------------------------------------------
__global__ __launch_bounds__(256) void k_head_gemm(
    const unsigned short* __restrict__ S,     // [BATCH][HID] bf16
    const unsigned short* __restrict__ W1bf,  // [OMID][HID] bf16
    const float* __restrict__ b1,             // [OMID]
    float* __restrict__ Mbuf)                 // [BATCH][OMID]
{
    const int lane = threadIdx.x & 31;
    const int wave = threadIdx.x >> 5;
    const int m_base = blockIdx.y * 128 + (wave >> 2) * 64;
    const int n_base = blockIdx.x * 64  + (wave & 3) * 16;
    const int col  = n_base + (lane & 15);
    const int half = lane >> 4;
    const int r    = lane & 15;

    v8f acc[4];
    const float b = b1[col];
#pragma unroll
    for (int mt = 0; mt < 4; ++mt) {
#pragma unroll
        for (int i = 0; i < 8; ++i) acc[mt][i] = b;
    }

    const unsigned short* pA[4];
#pragma unroll
    for (int mt = 0; mt < 4; ++mt)
        pA[mt] = S + (size_t)(m_base + mt * 16 + r) * HID + half * 16;
    const unsigned short* pB = W1bf + (size_t)(n_base + r) * HID + half * 16;

    v16bf a0[4], a1[4], b0, b1v;
#pragma unroll
    for (int mt = 0; mt < 4; ++mt) { a0[mt] = load_a_ptr(pA[mt]); pA[mt] += 32; }
    b0 = *(const v16bf*)pB; pB += 32;

#pragma unroll 1
    for (int k = 0; k < HID; k += 64) {
#pragma unroll
        for (int mt = 0; mt < 4; ++mt) { a1[mt] = load_a_ptr(pA[mt]); pA[mt] += 32; }
        b1v = *(const v16bf*)pB; pB += 32;
#pragma unroll
        for (int mt = 0; mt < 4; ++mt) acc[mt] = WMMA_BF16(a0[mt], b0, acc[mt]);
        if (k + 64 < HID) {
#pragma unroll
            for (int mt = 0; mt < 4; ++mt) { a0[mt] = load_a_ptr(pA[mt]); pA[mt] += 32; }
            b0 = *(const v16bf*)pB; pB += 32;
        }
#pragma unroll
        for (int mt = 0; mt < 4; ++mt) acc[mt] = WMMA_BF16(a1[mt], b1v, acc[mt]);
    }

#pragma unroll
    for (int mt = 0; mt < 4; ++mt) {
#pragma unroll
        for (int v = 0; v < 8; ++v) {
            const int row = m_base + mt * 16 + half * 8 + v;
            Mbuf[(size_t)row * OMID + col] = fmaxf(acc[mt][v], 0.0f);
        }
    }
}

// ---------------------------------------------------------------------------
// Per-row halting logic: one wave32 per batch row.
// ---------------------------------------------------------------------------
__global__ __launch_bounds__(256) void k_halt(
    const float* __restrict__ Mbuf,           // [BATCH][OMID]
    const unsigned short* __restrict__ S,     // [BATCH][HID] bf16
    const float* __restrict__ W2,             // [OMID]
    const float* __restrict__ b2,             // [1]
    const float* __restrict__ W_halt,         // [HID]
    const float* __restrict__ b_halt,         // [1]
    float* __restrict__ p_sum,                // [BATCH]
    int* __restrict__ active,                 // [BATCH]
    float* __restrict__ out_acc,              // [BATCH] == d_out
    int t)
{
    const int lane = threadIdx.x & 31;
    const int row  = blockIdx.x * 8 + (threadIdx.x >> 5);

    float s1 = 0.0f;  // relu(M) . W2
#pragma unroll
    for (int j = 0; j < OMID / 32; ++j) {
        const int c = lane + 32 * j;
        s1 += Mbuf[(size_t)row * OMID + c] * W2[c];
    }
    float s2 = 0.0f;  // state . W_halt
#pragma unroll
    for (int j = 0; j < HID / 32; ++j) {
        const int c = lane + 32 * j;
        s2 += bf2f(S[(size_t)row * HID + c]) * W_halt[c];
    }
#pragma unroll
    for (int off = 16; off > 0; off >>= 1) {
        s1 += __shfl_xor(s1, off, 32);
        s2 += __shfl_xor(s2, off, 32);
    }

    if (lane == 0) {
        const int   am = active[row];
        const float p  = p_sum[row];
        const float out = sigm(s1 + b2[0]);
        const float h   = sigm(s2 + b_halt[0]);
        const float p_new = p + h;
        const bool last = (t == MAX_ITER - 1);
        const bool fin  = am && ((p_new >= 1.0f - EPS) || last);
        const float halt_rec = fin ? (1.0f - p) : (am ? h : 0.0f);
        const float contrib  = am ? out * halt_rec : 0.0f;
        if (t == 0) out_acc[row] = contrib; else out_acc[row] += contrib;
        p_sum[row]  = fin ? 1.0f : (am ? p_new : p);
        active[row] = (am && !fin) ? 1 : 0;
    }
}

// ------------------------------ prep kernels -------------------------------
__global__ void k_prep_wcat(const float* __restrict__ Wxi, const float* __restrict__ Whi,
                            const float* __restrict__ Wxf, const float* __restrict__ Whf,
                            const float* __restrict__ Wxc, const float* __restrict__ Whc,
                            const float* __restrict__ Wxo, const float* __restrict__ Who,
                            unsigned short* __restrict__ Wcat) {
    const int idx = blockIdx.x * blockDim.x + threadIdx.x;
    if (idx >= 4 * HID * KA) return;
    const int rr = idx / KA, c = idx % KA;
    const int g = rr >> 10, n = rr & (HID - 1);
    const float* Wx = (g == 0) ? Wxi : (g == 1) ? Wxf : (g == 2) ? Wxc : Wxo;
    const float* Wh = (g == 0) ? Whi : (g == 1) ? Whf : (g == 2) ? Whc : Who;
    const float v = (c < IN) ? Wx[n * IN + c] : Wh[(size_t)n * HID + (c - IN)];
    Wcat[idx] = f2bf(v);
}

__global__ void k_prep_bcat(const float* __restrict__ bxi, const float* __restrict__ bhi,
                            const float* __restrict__ bxf, const float* __restrict__ bhf,
                            const float* __restrict__ bxc, const float* __restrict__ bhc,
                            const float* __restrict__ bxo, const float* __restrict__ bho,
                            float* __restrict__ bcat) {
    const int idx = blockIdx.x * blockDim.x + threadIdx.x;
    if (idx >= 4 * HID) return;
    const int g = idx >> 10, n = idx & (HID - 1);
    const float* bx = (g == 0) ? bxi : (g == 1) ? bxf : (g == 2) ? bxc : bxo;
    const float* bh = (g == 0) ? bhi : (g == 1) ? bhf : (g == 2) ? bhc : bho;
    bcat[idx] = bx[n] + bh[n];
}

__global__ void k_prep_w1(const float* __restrict__ W1, unsigned short* __restrict__ W1bf) {
    const int idx = blockIdx.x * blockDim.x + threadIdx.x;
    if (idx < OMID * HID) W1bf[idx] = f2bf(W1[idx]);
}

__global__ void k_prep_x(const float* __restrict__ x, unsigned short* __restrict__ Xbf) {
    const int idx = blockIdx.x * blockDim.x + threadIdx.x;
    if (idx < BATCH * IN) Xbf[idx] = f2bf(x[idx]);
}

__global__ void k_init_state(unsigned short* __restrict__ S0, float* __restrict__ Cell,
                             float* __restrict__ p_sum, int* __restrict__ active) {
    const int idx = blockIdx.x * blockDim.x + threadIdx.x;
    if (idx >= BATCH * HID) return;
    S0[idx] = 0;
    Cell[idx] = 0.0f;
    if (idx < BATCH) { p_sum[idx] = 0.0f; active[idx] = 1; }
}

// ---------------------------------------------------------------------------
extern "C" void kernel_launch(void* const* d_in, const int* in_sizes, int n_in,
                              void* d_out, int out_size, void* d_ws, size_t ws_size,
                              hipStream_t stream) {
    (void)in_sizes; (void)n_in; (void)out_size; (void)ws_size;
    const float* x      = (const float*)d_in[0];
    const float* Wxi    = (const float*)d_in[1];
    const float* bxi    = (const float*)d_in[2];
    const float* Whi    = (const float*)d_in[3];
    const float* bhi    = (const float*)d_in[4];
    const float* Wxf    = (const float*)d_in[5];
    const float* bxf    = (const float*)d_in[6];
    const float* Whf    = (const float*)d_in[7];
    const float* bhf    = (const float*)d_in[8];
    const float* Wxc    = (const float*)d_in[9];
    const float* bxc    = (const float*)d_in[10];
    const float* Whc    = (const float*)d_in[11];
    const float* bhc    = (const float*)d_in[12];
    const float* Wxo    = (const float*)d_in[13];
    const float* bxo    = (const float*)d_in[14];
    const float* Who    = (const float*)d_in[15];
    const float* bho    = (const float*)d_in[16];
    const float* W_halt = (const float*)d_in[17];
    const float* b_halt = (const float*)d_in[18];
    const float* W1     = (const float*)d_in[19];
    const float* b1     = (const float*)d_in[20];
    const float* W2     = (const float*)d_in[21];
    const float* b2     = (const float*)d_in[22];
    float* out = (float*)d_out;

    char* w = (char*)d_ws;
    size_t off = 0;
    auto alloc = [&](size_t bytes) -> void* {
        void* p = w + off;
        off = (off + bytes + 255) & ~(size_t)255;
        return p;
    };
    unsigned short* Xbf   = (unsigned short*)alloc((size_t)BATCH * IN * 2);
    unsigned short* S0    = (unsigned short*)alloc((size_t)BATCH * HID * 2);
    unsigned short* S1    = (unsigned short*)alloc((size_t)BATCH * HID * 2);
    unsigned short* Wcat  = (unsigned short*)alloc((size_t)4 * HID * KA * 2);
    unsigned short* W1bf  = (unsigned short*)alloc((size_t)OMID * HID * 2);
    float*          bcat  = (float*)alloc((size_t)4 * HID * 4);
    float*          Cell  = (float*)alloc((size_t)BATCH * HID * 4);
    float*          Mbuf  = (float*)alloc((size_t)BATCH * OMID * 4);
    float*          p_sum = (float*)alloc((size_t)BATCH * 4);
    int*            activ = (int*)alloc((size_t)BATCH * 4);

    // ---- prep (deterministic, every call) ----
    k_prep_wcat<<<(4 * HID * KA + 255) / 256, 256, 0, stream>>>(
        Wxi, Whi, Wxf, Whf, Wxc, Whc, Wxo, Who, Wcat);
    k_prep_bcat<<<(4 * HID + 255) / 256, 256, 0, stream>>>(
        bxi, bhi, bxf, bhf, bxc, bhc, bxo, bho, bcat);
    k_prep_w1<<<(OMID * HID + 255) / 256, 256, 0, stream>>>(W1, W1bf);
    k_prep_x<<<(BATCH * IN + 255) / 256, 256, 0, stream>>>(x, Xbf);
    k_init_state<<<(BATCH * HID + 255) / 256, 256, 0, stream>>>(S0, Cell, p_sum, activ);

    // ---- 8 ACT iterations, state ping-pong ----
    unsigned short* Sread = S0;
    unsigned short* Swrite = S1;
    for (int t = 0; t < MAX_ITER; ++t) {
        k_gates_lstm<<<dim3(HID / 64, BATCH / 128), 256, 0, stream>>>(
            Xbf, Sread, Swrite, Wcat, bcat, Cell, activ);
        k_head_gemm<<<dim3(OMID / 64, BATCH / 128), 256, 0, stream>>>(
            Swrite, W1bf, b1, Mbuf);
        k_halt<<<BATCH / 8, 256, 0, stream>>>(
            Mbuf, Swrite, W2, b2, W_halt, b_halt, p_sum, activ, out, t);
        unsigned short* tmp = Sread; Sread = Swrite; Swrite = tmp;
    }
}
